// OnlineEnVI_40905268527431
// MI455X (gfx1250) — matrix operand order repaired and checked by reading, not compiled, not run on hardware
//
#include <hip/hip_runtime.h>
#include <hip/hip_bf16.h>

#define D_  32
#define DY_ 32
#define M_  128
#define N_  128
#define B_  8
#define T_  48
#define JIT_F 1e-4f
#define LOG2PI_F 1.8378770664093453f

typedef __attribute__((ext_vector_type(16))) _Float16 v16h;
typedef __attribute__((ext_vector_type(8)))  float    v8f;

// ---------------------------------------------------------------------------
// WMMA fragment helpers (exact CDNA5 wave32 layouts from CDNA5 ISA)
// A (16x32 f16): lane<16 -> row=lane, vgpr v<4: k=2v+p, v>=4: k=16+2(v-4)+p
//                lane>=16 -> row=lane-16, v<4: k=8+2v+p, v>=4: k=24+2(v-4)+p
// B (32x16 f16): lane<16 -> col=lane ; lane>=16 -> col=lane-16 ; k = same mapping as A
// C/D (16x16 f32, 8 VGPRs): vgpr r -> row = r + (lane>=16 ? 8:0), col = lane&15
// ---------------------------------------------------------------------------
__device__ __forceinline__ void a_coord(int lane, int h, int& row, int& k) {
    int hi = lane >> 4; row = lane & 15;
    int v = h >> 1, p = h & 1;
    int base = (v < 4) ? (hi ? 8 : 0) : (hi ? 24 : 16);
    k = base + ((v & 3) << 1) + p;
}
__device__ __forceinline__ void b_coord(int lane, int h, int& k, int& col) {
    int hi = lane >> 4; col = lane & 15;
    int v = h >> 1, p = h & 1;
    k = ((v < 4) ? (hi ? 8 : 0) : (hi ? 24 : 16)) + ((v & 3) << 1) + p;
}

__device__ __forceinline__ v16h frag_a_f32(const float* p, int ld) {
    int lane = threadIdx.x & 31; v16h r;
#pragma unroll
    for (int h = 0; h < 16; ++h) { int row, k; a_coord(lane, h, row, k); r[h] = (_Float16)p[row * ld + k]; }
    return r;
}
__device__ __forceinline__ v16h frag_a_f16(const _Float16* p, int ld) {
    int lane = threadIdx.x & 31; v16h r;
#pragma unroll
    for (int h = 0; h < 16; ++h) { int row, k; a_coord(lane, h, row, k); r[h] = p[row * ld + k]; }
    return r;
}
__device__ __forceinline__ v16h frag_aT_f16(const _Float16* p, int ld) { // A[r,k] = p[k*ld+r]
    int lane = threadIdx.x & 31; v16h r;
#pragma unroll
    for (int h = 0; h < 16; ++h) { int row, k; a_coord(lane, h, row, k); r[h] = p[k * ld + row]; }
    return r;
}
__device__ __forceinline__ v16h frag_b_f16(const _Float16* p, int ld) { // B[k,c] = p[k*ld+c]
    int lane = threadIdx.x & 31; v16h r;
#pragma unroll
    for (int h = 0; h < 16; ++h) { int k, c; b_coord(lane, h, k, c); r[h] = p[k * ld + c]; }
    return r;
}
__device__ __forceinline__ v16h frag_bT_f32(const float* p, int ld) { // B[k,c] = p[c*ld+k]
    int lane = threadIdx.x & 31; v16h r;
#pragma unroll
    for (int h = 0; h < 16; ++h) { int k, c; b_coord(lane, h, k, c); r[h] = (_Float16)p[c * ld + k]; }
    return r;
}
__device__ __forceinline__ v16h frag_bT_f16_mask(const _Float16* p, int ld, int ncols) {
    int lane = threadIdx.x & 31; v16h r;
#pragma unroll
    for (int h = 0; h < 16; ++h) { int k, c; b_coord(lane, h, k, c);
        r[h] = (c < ncols) ? p[c * ld + k] : (_Float16)0.0f; }
    return r;
}
__device__ __forceinline__ v8f wmma16(v16h a, v16h b, v8f c) {
    return __builtin_amdgcn_wmma_f32_16x16x32_f16(false, a, false, b, (short)0, c, false, false);
}
__device__ __forceinline__ float block_reduce256(float v) {
    __shared__ float red[256];
    red[threadIdx.x] = v; __syncthreads();
    for (int s = 128; s > 0; s >>= 1) { if ((int)threadIdx.x < s) red[threadIdx.x] += red[threadIdx.x + s]; __syncthreads(); }
    return red[0];
}

// ======================= precompute kernels ================================
__global__ void k_init(float* sc) { sc[0] = 0.0f; sc[1] = 0.0f; }

__global__ void k_zs(const float* __restrict__ ips, const float* __restrict__ ls, float* zs) {
    int i = blockIdx.x * 256 + threadIdx.x;            // D*M*D
    int d = i / (M_ * D_);
    zs[i] = ips[i] / ls[d];
}

__global__ void k_kzz(const float* __restrict__ zs, const float* __restrict__ osc, float* Kzz) {
    int i = blockIdx.x * 256 + threadIdx.x;            // D*M*M
    int d = i / (M_ * M_), r = (i / M_) % M_, c = i % M_;
    const float* a = zs + (size_t)d * M_ * D_ + (size_t)r * D_;
    const float* b = zs + (size_t)d * M_ * D_ + (size_t)c * D_;
    float s = 0.f;
    for (int k = 0; k < D_; ++k) { float df = a[k] - b[k]; s += df * df; }
    Kzz[i] = osc[d] * __expf(-0.5f * s) + ((r == c) ? JIT_F : 0.f);
}

__global__ void k_chol128(const float* __restrict__ Kzz, float* __restrict__ Lzz) {
    __shared__ float sh[M_ * M_];                      // 64KB of the 320KB WGP LDS
    int d = blockIdx.x, t = threadIdx.x;               // 128 threads, thread t owns row t
    for (int i = 0; i < M_; ++i) sh[i * M_ + t] = Kzz[(size_t)d * M_ * M_ + i * M_ + t];
    __syncthreads();
    for (int j = 0; j < M_; ++j) {
        if (t == j) sh[j * M_ + j] = sqrtf(fmaxf(sh[j * M_ + j], 1e-20f));
        __syncthreads();
        if (t > j) sh[t * M_ + j] /= sh[j * M_ + j];
        __syncthreads();
        if (t > j) { float l = sh[t * M_ + j];
            for (int k = j + 1; k <= t; ++k) sh[t * M_ + k] -= l * sh[k * M_ + j]; }
        __syncthreads();
    }
    for (int i = 0; i < M_; ++i)
        Lzz[(size_t)d * M_ * M_ + i * M_ + t] = (t <= i) ? sh[i * M_ + t] : 0.0f;
}

__global__ void k_inv128(const float* __restrict__ Lzz, float* Kinv, _Float16* Kinvh) {
    __shared__ float L[M_ * M_];
    int d = blockIdx.x, c = threadIdx.x;               // thread c solves column c of K^-1
    for (int i = 0; i < M_; ++i) L[i * M_ + c] = Lzz[(size_t)d * M_ * M_ + i * M_ + c];
    __syncthreads();
    float y[M_];
    for (int i = 0; i < M_; ++i) {                     // L y = e_c
        float s = (i == c) ? 1.0f : 0.0f;
        for (int k = 0; k < i; ++k) s -= L[i * M_ + k] * y[k];
        y[i] = s / L[i * M_ + i];
    }
    for (int i = M_ - 1; i >= 0; --i) {                // L^T x = y (overwrite)
        float s = y[i];
        for (int k = i + 1; k < M_; ++k) s -= L[k * M_ + i] * y[k];
        y[i] = s / L[i * M_ + i];
    }
    for (int i = 0; i < M_; ++i) {
        Kinv[(size_t)d * M_ * M_ + i * M_ + c]  = y[i];
        Kinvh[(size_t)d * M_ * M_ + i * M_ + c] = (_Float16)y[i];
    }
}

// U[n,d,:] = q_mu[d,:] + eps_u[n,d,:] @ q_L[d]^T     (WMMA, f32->f16 convert-on-load)
__global__ void k_ugemm(const float* __restrict__ eps_u, const float* __restrict__ qL,
                        const float* __restrict__ qmu, float* U) {
    int tile = blockIdx.x * 8 + (threadIdx.x >> 5);    // D*64 tiles, 8 waves/block
    int lane = threadIdx.x & 31;
    int d = tile >> 6, rem = tile & 63, nt = rem >> 3, mt = rem & 7;
    v8f acc = {};
    for (int kk = 0; kk < M_; kk += 32) {
        v16h a  = frag_a_f32(eps_u + (size_t)(nt * 16) * D_ * M_ + (size_t)d * M_ + kk, D_ * M_);
        v16h bf = frag_bT_f32(qL + (size_t)d * M_ * M_ + (size_t)(mt * 16) * M_ + kk, M_);
        acc = wmma16(a, bf, acc);
    }
    int hi = lane >> 4, col = lane & 15;
#pragma unroll
    for (int r = 0; r < 8; ++r) {
        int n = nt * 16 + r + (hi ? 8 : 0), m = mt * 16 + col;
        U[((size_t)n * D_ + d) * M_ + m] = acc[r] + qmu[d * M_ + m];
    }
}

// A[n,d,:] = U[n,d,:] @ Kinv[d]   (Kinv symmetric)   (WMMA)
__global__ void k_agemm(const float* __restrict__ U, const _Float16* __restrict__ Kinvh,
                        float* Acoef) {
    int tile = blockIdx.x * 8 + (threadIdx.x >> 5);
    int lane = threadIdx.x & 31;
    int d = tile >> 6, rem = tile & 63, nt = rem >> 3, mt = rem & 7;
    v8f acc = {};
    for (int kk = 0; kk < M_; kk += 32) {
        v16h a  = frag_a_f32(U + (size_t)(nt * 16) * D_ * M_ + (size_t)d * M_ + kk, D_ * M_);
        v16h bf = frag_b_f16(Kinvh + (size_t)d * M_ * M_ + (size_t)kk * M_ + mt * 16, M_);
        acc = wmma16(a, bf, acc);
    }
    int hi = lane >> 4, col = lane & 15;
#pragma unroll
    for (int r = 0; r < 8; ++r) {
        int n = nt * 16 + r + (hi ? 8 : 0), m = mt * 16 + col;
        Acoef[((size_t)n * D_ + d) * M_ + m] = acc[r];
    }
}

// KL: 0.5*tr(Kinv qL qL^T) accumulated per (d,r,c) lower-tri entry
__global__ void k_kl_trace(const float* __restrict__ Kinv, const float* __restrict__ qL,
                           float* klsum) {
    int i = blockIdx.x * 256 + threadIdx.x;            // D*M*M
    int d = i / (M_ * M_), r = (i / M_) % M_, c = i % M_;
    float v = 0.f;
    if (r >= c) {
        const float* ki = Kinv + (size_t)d * M_ * M_ + (size_t)r * M_;
        const float* qc = qL + (size_t)d * M_ * M_ + c;
        float s = 0.f;
        for (int k = c; k < M_; ++k) s += ki[k] * qc[(size_t)k * M_];
        v = 0.5f * s * qL[i];
    }
    v = block_reduce256(v);
    if (threadIdx.x == 0) atomicAdd(klsum, v);
}

// 0.5*mah + 0.5*logdetK - 0.5*logdetS, per (d,m)
__global__ void k_kl_rest(const float* __restrict__ Kinv, const float* __restrict__ qmu,
                          const float* __restrict__ Lzz, const float* __restrict__ qL,
                          float* klsum) {
    int i = blockIdx.x * 256 + threadIdx.x;            // D*M
    int d = i / M_, m = i % M_;
    const float* ki = Kinv + (size_t)d * M_ * M_ + (size_t)m * M_;
    const float* mu = qmu + (size_t)d * M_;
    float s = 0.f;
    for (int k = 0; k < M_; ++k) s += ki[k] * mu[k];
    float v = 0.5f * mu[m] * s;
    v += __logf(fmaxf(Lzz[(size_t)d * M_ * M_ + m * M_ + m], 1e-30f));
    v -= __logf(fmaxf(fabsf(qL[(size_t)d * M_ * M_ + m * M_ + m]), 1e-30f));
    v = block_reduce256(v);
    if (threadIdx.x == 0) atomicAdd(klsum, v);
}

__global__ void k_taper(float* taper) {
    int i = blockIdx.x * 256 + threadIdx.x;
    if (i >= D_ * D_) return;
    int r = i / D_, c = i % D_;
    float ad = fabsf((float)(r - c));
    float dist = fminf(ad, (float)D_ - ad);
    float z = dist / 5.0f, zsafe = fmaxf(z, 1e-12f);
    float z2 = z * z, z3 = z2 * z, z4 = z2 * z2, z5 = z4 * z;
    float g1 = 1.f - (5.f / 3.f) * z2 + (5.f / 8.f) * z3 + 0.5f * z4 - 0.25f * z5;
    float g2 = 4.f - 5.f * z + (5.f / 3.f) * z2 + (5.f / 8.f) * z3 - 0.5f * z4
             + (1.f / 12.f) * z5 - 2.f / (3.f * zsafe);
    taper[i] = (z < 1.f) ? g1 : ((z < 2.f) ? g2 : 0.f);
}

__global__ void k_copy(const float* __restrict__ src, float* __restrict__ dst) {
    size_t i = (size_t)blockIdx.x * 256 + threadIdx.x;
    dst[i] = src[i];
}

// ========================= per-step kernels ================================
// S1: Kxz (f32+f16), Kxx, mean = Kxz @ A   — one block per (n,d)
__global__ void k_kxz(const float* __restrict__ xcur, const float* __restrict__ zs,
                      const float* __restrict__ osc, const float* __restrict__ ls,
                      const float* __restrict__ Acoef, float* Kxz, _Float16* Kxzh,
                      float* Kxx, float* meanv) {
    int nd = blockIdx.x, n = nd / D_, d = nd % D_, t = threadIdx.x;
    __shared__ float zsh[M_ * D_];                     // 16KB
    __shared__ float xq[B_ * D_];
    __shared__ float z2[M_], x2[B_];
    __shared__ float krow[B_ * M_];                    // 4KB
    for (int i = t; i < M_ * D_; i += 256) zsh[i] = zs[(size_t)d * M_ * D_ + i];
    for (int i = t; i < B_ * D_; i += 256) {
        int b = i / D_, k = i % D_;
        xq[i] = xcur[(((size_t)n * D_ + d) * B_ + b) * D_ + k] / ls[d];
    }
    __syncthreads();
    if (t < M_) { float s = 0.f; for (int k = 0; k < D_; ++k) { float v = zsh[t * D_ + k]; s += v * v; } z2[t] = s; }
    else if (t < M_ + B_) { int b = t - M_; float s = 0.f; for (int k = 0; k < D_; ++k) { float v = xq[b * D_ + k]; s += v * v; } x2[b] = s; }
    __syncthreads();
    float oscd = osc[d];
    for (int i = t; i < B_ * M_; i += 256) {
        int b = i / M_, m = i % M_;
        float dot = 0.f;
        for (int k = 0; k < D_; ++k) dot += xq[b * D_ + k] * zsh[m * D_ + k];
        float d2 = fmaxf(x2[b] + z2[m] - 2.f * dot, 0.f);
        float kv = oscd * __expf(-0.5f * d2);
        krow[i] = kv;
        Kxz[(size_t)nd * B_ * M_ + i]  = kv;
        Kxzh[(size_t)nd * B_ * M_ + i] = (_Float16)kv;
    }
    for (int i = t; i < B_ * B_; i += 256) {
        int b = i / B_, b2 = i % B_;
        float dot = 0.f;
        for (int k = 0; k < D_; ++k) dot += xq[b * D_ + k] * xq[b2 * D_ + k];
        float d2 = fmaxf(x2[b] + x2[b2] - 2.f * dot, 0.f);
        Kxx[(size_t)nd * B_ * B_ + i] = oscd * __expf(-0.5f * d2);
    }
    __syncthreads();
    if (t < B_) {
        float s = 0.f;
        const float* Ap = Acoef + ((size_t)n * D_ + d) * M_;
        for (int m = 0; m < M_; ++m) s += krow[t * M_ + m] * Ap[m];
        meanv[(size_t)nd * B_ + t] = s;
    }
}

// S2: dominant batched GEMM  V[nd] = Kinv[d] (128x128) @ Kxz[nd]^T (128x8)  — WMMA
__global__ void k_vgemm(const _Float16* __restrict__ Kinvh, const _Float16* __restrict__ Kxzh,
                        float* __restrict__ V) {
    int nd = blockIdx.x, d = nd % D_;
    int wave = threadIdx.x >> 5, lane = threadIdx.x & 31;  // 8 waves = 8 row tiles of M
    const _Float16* Ad = Kinvh + (size_t)d * M_ * M_ + (size_t)(wave * 16) * M_;
    const _Float16* Bt = Kxzh + (size_t)nd * B_ * M_;
    v8f acc = {};
    for (int kk = 0; kk < M_; kk += 32) {
        __builtin_prefetch(Ad + (kk + 32 < M_ ? kk + 32 : kk), 0, 1);   // global_prefetch_b8
        v16h a  = frag_a_f16(Ad + kk, M_);
        v16h bf = frag_bT_f16_mask(Bt + kk, M_, B_);
        acc = wmma16(a, bf, acc);
    }
    int hi = lane >> 4, col = lane & 15;
    if (col < B_) {
#pragma unroll
        for (int r = 0; r < 8; ++r) {
            int row = wave * 16 + r + (hi ? 8 : 0);
            V[((size_t)nd * M_ + row) * B_ + col] = acc[r];
        }
    }
}

// S3: cov = Kxx - Kxz V + JIT*I ; 8x8 Cholesky ; f = mean + Lc ef ; X assembly
__global__ void k_cov(const float* __restrict__ Kxx, const float* __restrict__ Kxz,
                      const float* __restrict__ V, const float* __restrict__ meanv,
                      const float* __restrict__ eps_f, const float* __restrict__ eps_q,
                      const float* __restrict__ q_noise, float* X, int t) {
    int nd = blockIdx.x, n = nd / D_, d = nd % D_, tid = threadIdx.x;   // 64 threads
    __shared__ float cov[B_ * B_];
    int b = tid / B_, b2 = tid % B_;
    {
        const float* kx = Kxz + (size_t)nd * B_ * M_ + (size_t)b * M_;
        const float* vp = V + (size_t)nd * M_ * B_ + b2;
        float s = 0.f;
        for (int m = 0; m < M_; ++m) s += kx[m] * vp[(size_t)m * B_];
        cov[tid] = Kxx[(size_t)nd * B_ * B_ + tid] - s + ((b == b2) ? JIT_F : 0.f);
    }
    __syncthreads();
    if (tid == 0) {                                    // tiny 8x8 Cholesky
        for (int j = 0; j < B_; ++j) {
            float v = cov[j * B_ + j];
            for (int k = 0; k < j; ++k) v -= cov[j * B_ + k] * cov[j * B_ + k];
            v = sqrtf(fmaxf(v, 1e-20f));
            cov[j * B_ + j] = v;
            for (int i = j + 1; i < B_; ++i) {
                float s2 = cov[i * B_ + j];
                for (int k = 0; k < j; ++k) s2 -= cov[i * B_ + k] * cov[j * B_ + k];
                cov[i * B_ + j] = s2 / v;
            }
        }
    }
    __syncthreads();
    if (tid < B_) {
        int bb = tid;
        const float* ef = eps_f + (((size_t)t * N_ + n) * D_ + d) * B_;
        float f = meanv[(size_t)nd * B_ + bb];
        for (int c = 0; c <= bb; ++c) f += cov[bb * B_ + c] * ef[c];
        float eq = eps_q[(((size_t)t * B_ + bb) * N_ + n) * D_ + d];
        X[((size_t)bb * N_ + n) * D_ + d] = f + eq * sqrtf(q_noise[d]);
    }
}

__global__ void k_xmean(const float* __restrict__ X, float* Xm) {
    int t = threadIdx.x, b = t / D_, d = t % D_;       // 256 = B*D
    float s = 0.f;
    for (int n = 0; n < N_; ++n) s += X[((size_t)b * N_ + n) * D_ + d];
    Xm[t] = s / (float)N_;
}

__global__ void k_center(const float* __restrict__ X, const float* __restrict__ Xm,
                         _Float16* Xch) {
    int i = blockIdx.x * 256 + threadIdx.x;            // B*N*D
    int d = i % D_, b = i / (N_ * D_);
    Xch[i] = (_Float16)(X[i] - Xm[b * D_ + d]);
}

// S5: P[b] = taper .* (Xc^T Xc) / (N-1)   — WMMA, 4 tiles per b (4 waves)
__global__ void k_pcov(const _Float16* __restrict__ Xch, const float* __restrict__ taper,
                       float* P) {
    int b = blockIdx.x;
    int wave = threadIdx.x >> 5, lane = threadIdx.x & 31;
    int ti = wave >> 1, tj = wave & 1;
    const _Float16* base = Xch + (size_t)b * N_ * D_;
    v8f acc = {};
    for (int kk = 0; kk < N_; kk += 32) {
        v16h a  = frag_aT_f16(base + (size_t)kk * D_ + ti * 16, D_);  // Xc^T
        v16h bf = frag_b_f16(base + (size_t)kk * D_ + tj * 16, D_);   // Xc
        acc = wmma16(a, bf, acc);
    }
    int hi = lane >> 4, col = lane & 15;
#pragma unroll
    for (int r = 0; r < 8; ++r) {
        int row = ti * 16 + r + (hi ? 8 : 0), cg = tj * 16 + col;
        P[((size_t)b * D_ + row) * D_ + cg] =
            acc[r] * (1.0f / (float)(N_ - 1)) * taper[row * D_ + cg];
    }
}

// S6: S = H P H^T + R; chol; ll; gain K; P_post diag -> filtered_var
__global__ void k_update(const float* __restrict__ P, const float* __restrict__ H,
                         const float* __restrict__ r_noise, const float* __restrict__ y,
                         const float* __restrict__ Xm, float* Kg, float* ll,
                         float* fv_out, int t) {
    int b = blockIdx.x, tid = threadIdx.x;             // 128 threads
    __shared__ float Ps[D_ * D_], HP[DY_ * D_], Ss[DY_ * DY_], PHt[D_ * DY_];
    __shared__ float resid[DY_];
    for (int i = tid; i < D_ * D_; i += 128) Ps[i] = P[(size_t)b * D_ * D_ + i];
    __syncthreads();
    for (int i = tid; i < DY_ * D_; i += 128) { int c = i / D_, j = i % D_;
        float s = 0.f; for (int k = 0; k < D_; ++k) s += H[c * D_ + k] * Ps[k * D_ + j]; HP[i] = s; }
    __syncthreads();
    for (int i = tid; i < DY_ * DY_; i += 128) { int r = i / DY_, c = i % DY_;
        float s = (r == c) ? r_noise[r] : 0.f;
        for (int l = 0; l < D_; ++l) s += HP[r * D_ + l] * H[c * D_ + l]; Ss[i] = s; }
    for (int i = tid; i < D_ * DY_; i += 128) { int r = i / DY_, c = i % DY_;
        float s = 0.f; for (int j = 0; j < D_; ++j) s += Ps[r * D_ + j] * H[c * D_ + j]; PHt[i] = s; }
    __syncthreads();
    for (int j = 0; j < DY_; ++j) {                    // cooperative 32x32 Cholesky
        if (tid == j) Ss[j * DY_ + j] = sqrtf(fmaxf(Ss[j * DY_ + j], 1e-20f));
        __syncthreads();
        if (tid < DY_ && tid > j) Ss[tid * DY_ + j] /= Ss[j * DY_ + j];
        __syncthreads();
        if (tid < DY_ && tid > j) { float l = Ss[tid * DY_ + j];
            for (int k = j + 1; k <= tid; ++k) Ss[tid * DY_ + k] -= l * Ss[k * DY_ + j]; }
        __syncthreads();
    }
    if (tid < DY_) {
        float s = 0.f;
        for (int dd = 0; dd < D_; ++dd) s += Xm[b * D_ + dd] * H[tid * D_ + dd];
        resid[tid] = y[((size_t)b * T_ + t) * DY_ + tid] - s;
    }
    __syncthreads();
    if (tid == 0) {
        float ld = 0.f;
        for (int j = 0; j < DY_; ++j) ld += 2.f * __logf(fmaxf(Ss[j * DY_ + j], 1e-30f));
        float tmp[DY_];
        for (int i = 0; i < DY_; ++i) { float s = resid[i];
            for (int k = 0; k < i; ++k) s -= Ss[i * DY_ + k] * tmp[k]; tmp[i] = s / Ss[i * DY_ + i]; }
        for (int i = DY_ - 1; i >= 0; --i) { float s = tmp[i];
            for (int k = i + 1; k < DY_; ++k) s -= Ss[k * DY_ + i] * tmp[k]; tmp[i] = s / Ss[i * DY_ + i]; }
        float quad = 0.f;
        for (int i = 0; i < DY_; ++i) quad += resid[i] * tmp[i];
        atomicAdd(ll, -0.5f * ((float)DY_ * LOG2PI_F + ld + quad));
    }
    __syncthreads();
    if (tid < D_) {                                    // gain row tid: solve S x = PHt[tid,:]
        float xv[DY_];
        for (int i = 0; i < DY_; ++i) { float s = PHt[tid * DY_ + i];
            for (int k = 0; k < i; ++k) s -= Ss[i * DY_ + k] * xv[k]; xv[i] = s / Ss[i * DY_ + i]; }
        for (int i = DY_ - 1; i >= 0; --i) { float s = xv[i];
            for (int k = i + 1; k < DY_; ++k) s -= Ss[k * DY_ + i] * xv[k]; xv[i] = s / Ss[i * DY_ + i]; }
        float pd = Ps[tid * D_ + tid];
        for (int c = 0; c < DY_; ++c) {
            Kg[((size_t)b * D_ + tid) * DY_ + c] = xv[c];
            pd -= xv[c] * HP[c * D_ + tid];
        }
        fv_out[((size_t)b * T_ + t) * D_ + tid] = pd;  // filtered_var (B,T,D)
    }
}

// S7: fused  E = y_pert - X@Ht ;  X += E @ K^T   — two WMMA phases per b
__global__ void k_xpost(const float* __restrict__ H, const float* __restrict__ Kg,
                        const float* __restrict__ y, const float* __restrict__ eps_r,
                        const float* __restrict__ r_noise, float* X, int t) {
    int b = blockIdx.x;
    int wave = threadIdx.x >> 5, lane = threadIdx.x & 31;  // 8 waves = 8 row tiles of N
    __shared__ float E[N_ * DY_];                      // 16KB
    float* Xb = X + (size_t)b * N_ * D_;
#pragma unroll
    for (int ct = 0; ct < 2; ++ct) {                   // K = D_ = 32: single WMMA
        v8f acc = {};
        v16h a  = frag_a_f32(Xb + (size_t)(wave * 16) * D_, D_);
        v16h bf = frag_bT_f32(H + (size_t)(ct * 16) * D_, D_);
        acc = wmma16(a, bf, acc);
        int hi = lane >> 4, col = lane & 15;
#pragma unroll
        for (int r = 0; r < 8; ++r) {
            int n = wave * 16 + r + (hi ? 8 : 0), c = ct * 16 + col;
            float er = eps_r[(((size_t)t * B_ + b) * N_ + n) * DY_ + c];
            float yp = y[((size_t)b * T_ + t) * DY_ + c] + er * sqrtf(r_noise[c]);
            E[n * DY_ + c] = yp - acc[r];
        }
    }
    __syncthreads();
#pragma unroll
    for (int ct = 0; ct < 2; ++ct) {                   // K = DY_ = 32: single WMMA
        v8f acc = {};
        v16h a  = frag_a_f32(E + (size_t)(wave * 16) * DY_, DY_);
        v16h bf = frag_bT_f32(Kg + (size_t)b * D_ * DY_ + (size_t)(ct * 16) * DY_, DY_);
        acc = wmma16(a, bf, acc);
        int hi = lane >> 4, col = lane & 15;
#pragma unroll
        for (int r = 0; r < 8; ++r) {
            int n = wave * 16 + r + (hi ? 8 : 0), dc = ct * 16 + col;
            Xb[(size_t)n * D_ + dc] += acc[r];
        }
    }
}

__global__ void k_postmean(const float* __restrict__ X, float* Xmp, float* fm_out, int t) {
    int i = threadIdx.x, b = i / D_, d = i % D_;       // 256 = B*D
    float s = 0.f;
    for (int n = 0; n < N_; ++n) s += X[((size_t)b * N_ + n) * D_ + d];
    s /= (float)N_;
    Xmp[i] = s;
    fm_out[((size_t)b * T_ + t) * D_ + d] = s;         // filtered_mean (B,T,D)
}

__global__ void k_broadcast(const float* __restrict__ Xmp, float* __restrict__ xcur) {
    size_t i = (size_t)blockIdx.x * 256 + threadIdx.x; // N*D*B*D
    int k = (int)(i & 31), b = (int)((i >> 5) & 7);
    xcur[i] = Xmp[b * D_ + k];
}

__global__ void k_finalize(const float* __restrict__ sc, float* out0) {
    float kl = sc[1] - 0.5f * (float)(D_ * M_);
    out0[0] = sc[0] - kl / 100000.0f;                  // elbo = ll - kl/BETA
}

// ============================== host side ==================================
extern "C" void kernel_launch(void* const* d_in, const int* in_sizes, int n_in,
                              void* d_out, int out_size, void* d_ws, size_t ws_size,
                              hipStream_t stream) {
    (void)in_sizes; (void)n_in; (void)out_size; (void)ws_size;
    const float* ips    = (const float*)d_in[0];
    const float* H      = (const float*)d_in[1];
    const float* ls     = (const float*)d_in[2];
    const float* osc    = (const float*)d_in[3];
    const float* qmu    = (const float*)d_in[4];
    const float* qL     = (const float*)d_in[5];
    const float* qnoise = (const float*)d_in[6];
    const float* rnoise = (const float*)d_in[7];
    const float* x0     = (const float*)d_in[8];
    const float* y      = (const float*)d_in[9];
    const float* eps_u  = (const float*)d_in[10];
    const float* eps_f  = (const float*)d_in[11];
    const float* eps_q  = (const float*)d_in[12];
    const float* eps_r  = (const float*)d_in[13];

    char* w = (char*)d_ws;
    auto grab = [&](size_t bytes) { char* p = w; w += (bytes + 255) & ~(size_t)255; return p; };
    float*     Kzz   = (float*)grab((size_t)D_ * M_ * M_ * 4);
    float*     Lzz   = (float*)grab((size_t)D_ * M_ * M_ * 4);
    float*     Kinv  = (float*)grab((size_t)D_ * M_ * M_ * 4);
    float*     zs    = (float*)grab((size_t)D_ * M_ * D_ * 4);
    float*     U     = (float*)grab((size_t)N_ * D_ * M_ * 4);
    float*     Acoef = (float*)grab((size_t)N_ * D_ * M_ * 4);
    float*     Kxz   = (float*)grab((size_t)N_ * D_ * B_ * M_ * 4);
    float*     V     = (float*)grab((size_t)N_ * D_ * M_ * B_ * 4);
    float*     Kxx   = (float*)grab((size_t)N_ * D_ * B_ * B_ * 4);
    float*     meanv = (float*)grab((size_t)N_ * D_ * B_ * 4);
    float*     X     = (float*)grab((size_t)B_ * N_ * D_ * 4);
    float*     Xm    = (float*)grab((size_t)B_ * D_ * 4);
    float*     P     = (float*)grab((size_t)B_ * D_ * D_ * 4);
    float*     Kg    = (float*)grab((size_t)B_ * D_ * DY_ * 4);
    float*     Xmp   = (float*)grab((size_t)B_ * D_ * 4);
    float*     taper = (float*)grab((size_t)D_ * D_ * 4);
    float*     sc    = (float*)grab(64);               // [0]=ll, [1]=klsum
    _Float16*  Kinvh = (_Float16*)grab((size_t)D_ * M_ * M_ * 2);
    _Float16*  Kxzh  = (_Float16*)grab((size_t)N_ * D_ * B_ * M_ * 2);
    _Float16*  Xch   = (_Float16*)grab((size_t)B_ * N_ * D_ * 2);

    float* out0 = (float*)d_out;
    float* xcur = out0 + 1;                            // x_final slot doubles as state
    float* fm   = out0 + 1 + (size_t)N_ * D_ * B_ * D_;
    float* fv   = fm + (size_t)B_ * T_ * D_;

    // ---- precompute ----
    k_init<<<1, 1, 0, stream>>>(sc);
    k_zs<<<(D_ * M_ * D_) / 256, 256, 0, stream>>>(ips, ls, zs);
    k_kzz<<<(D_ * M_ * M_) / 256, 256, 0, stream>>>(zs, osc, Kzz);
    k_chol128<<<D_, 128, 0, stream>>>(Kzz, Lzz);
    k_inv128<<<D_, 128, 0, stream>>>(Lzz, Kinv, Kinvh);
    k_ugemm<<<(D_ * 64) / 8, 256, 0, stream>>>(eps_u, qL, qmu, U);
    k_agemm<<<(D_ * 64) / 8, 256, 0, stream>>>(U, Kinvh, Acoef);
    k_kl_trace<<<(D_ * M_ * M_) / 256, 256, 0, stream>>>(Kinv, qL, sc + 1);
    k_kl_rest<<<(D_ * M_) / 256, 256, 0, stream>>>(Kinv, qmu, Lzz, qL, sc + 1);
    k_taper<<<(D_ * D_ + 255) / 256, 256, 0, stream>>>(taper);
    k_copy<<<(N_ * D_ * B_ * D_) / 256, 256, 0, stream>>>(x0, xcur);

    // ---- sequential filter scan ----
    for (int t = 0; t < T_; ++t) {
        k_kxz<<<N_ * D_, 256, 0, stream>>>(xcur, zs, osc, ls, Acoef, Kxz, Kxzh, Kxx, meanv);
        k_vgemm<<<N_ * D_, 256, 0, stream>>>(Kinvh, Kxzh, V);
        k_cov<<<N_ * D_, 64, 0, stream>>>(Kxx, Kxz, V, meanv, eps_f, eps_q, qnoise, X, t);
        k_xmean<<<1, 256, 0, stream>>>(X, Xm);
        k_center<<<(B_ * N_ * D_) / 256, 256, 0, stream>>>(X, Xm, Xch);
        k_pcov<<<B_, 128, 0, stream>>>(Xch, taper, P);
        k_update<<<B_, 128, 0, stream>>>(P, H, rnoise, y, Xm, Kg, sc, fv, t);
        k_xpost<<<B_, 256, 0, stream>>>(H, Kg, y, eps_r, rnoise, X, t);
        k_postmean<<<1, 256, 0, stream>>>(X, Xmp, fm, t);
        k_broadcast<<<(N_ * D_ * B_ * D_) / 256, 256, 0, stream>>>(Xmp, xcur);
    }
    k_finalize<<<1, 1, 0, stream>>>(sc, out0);
}
